// Qwen3MoeBlock_39642548142374
// MI455X (gfx1250) — matrix operand order repaired and checked by reading, not compiled, not run on hardware
//
#include <hip/hip_runtime.h>
#include <math.h>

// Problem constants (B=2, S=1024 -> 2048 tokens)
#define NTOK 2048
#define DDIM 2048
#define NEXP 8
#define FDIM 768
#define CAP  2048
#define LST  40   // LDS row stride in bf16 elements (64B data + 16B pad per row)

typedef __attribute__((ext_vector_type(16))) __bf16 v16bf;
typedef __attribute__((ext_vector_type(8)))  __bf16 v8bf;
typedef __attribute__((ext_vector_type(8)))  float  v8f;
typedef __attribute__((ext_vector_type(4)))  unsigned int u32x4;
typedef __attribute__((ext_vector_type(8)))  int i32x8;
typedef __attribute__((ext_vector_type(4)))  int i32x4;

union ABf16 { v16bf v16; v8bf v8[2]; };

__device__ __forceinline__ v8f wmma_bf16(const ABf16& a, const ABf16& b, v8f c) {
  return __builtin_amdgcn_wmma_f32_16x16x32_bf16(false, a.v16, false, b.v16,
                                                 (short)0, c, false, false);
}

// Convert 16 contiguous f32 -> 16 bf16, write 32B (two uint4) to LDS.
__device__ __forceinline__ void cvt16_f32_bf16(const float* __restrict__ src,
                                               __bf16* __restrict__ dst) {
  const float4* s4 = (const float4*)src;
  float4 a0 = s4[0], a1 = s4[1], a2 = s4[2], a3 = s4[3];
  __bf16 t[16] __attribute__((aligned(16)));
  t[ 0]=(__bf16)a0.x; t[ 1]=(__bf16)a0.y; t[ 2]=(__bf16)a0.z; t[ 3]=(__bf16)a0.w;
  t[ 4]=(__bf16)a1.x; t[ 5]=(__bf16)a1.y; t[ 6]=(__bf16)a1.z; t[ 7]=(__bf16)a1.w;
  t[ 8]=(__bf16)a2.x; t[ 9]=(__bf16)a2.y; t[10]=(__bf16)a2.z; t[11]=(__bf16)a2.w;
  t[12]=(__bf16)a3.x; t[13]=(__bf16)a3.y; t[14]=(__bf16)a3.z; t[15]=(__bf16)a3.w;
  *(uint4*)dst       = *(const uint4*)&t[0];
  *(uint4*)(dst + 8) = *(const uint4*)&t[8];
}

__device__ __forceinline__ v8f vzero8() {
  v8f z;
#pragma unroll
  for (int i = 0; i < 8; ++i) z[i] = 0.0f;
  return z;
}

// Load an A fragment (16x32 bf16) for this lane from a padded LDS tile.
// Lane-half selects K {base..base+7, base+16..base+23} per the ISA layout.
__device__ __forceinline__ ABf16 load_afrag(const __bf16* rowbase, int hi) {
  ABf16 a;
  a.v8[0] = *(const v8bf*)(rowbase + hi * 8);
  a.v8[1] = *(const v8bf*)(rowbase + hi * 8 + 16);
  return a;
}
// Load a B fragment (32x16 bf16): lane = N column, K {base..base+7, base+16..base+23} per lane-half.
__device__ __forceinline__ ABf16 load_bfrag(const __bf16* rowbase, int hi) {
  ABf16 b;
  b.v8[0] = *(const v8bf*)(rowbase + hi * 8);
  b.v8[1] = *(const v8bf*)(rowbase + hi * 8 + 16);
  return b;
}

// ---------------------------------------------------------------- K0: init
__global__ void moe_k0_init(int* counts) {
  if (threadIdx.x < NEXP) counts[threadIdx.x] = 0;
}

// ---------------------------------------------------------------- K1: router
__global__ __launch_bounds__(256) void moe_k1_router(
    const float* __restrict__ x, const float* __restrict__ wr,
    int* __restrict__ counts, int* __restrict__ tokl,
    int* __restrict__ slotl, float* __restrict__ wtl) {
  const int lane = threadIdx.x & 31;
  const int wv   = threadIdx.x >> 5;
  const int t    = blockIdx.x * 8 + wv;
  const float4* x4  = (const float4*)(x + (size_t)t * DDIM);
  const float4* wr4 = (const float4*)wr;

  float acc[NEXP];
#pragma unroll
  for (int e = 0; e < NEXP; ++e) acc[e] = 0.0f;

#pragma unroll 4
  for (int it = 0; it < (DDIM / 4) / 32; ++it) {
    float4 xv = x4[it * 32 + lane];
#pragma unroll
    for (int e = 0; e < NEXP; ++e) {
      float4 wvv = wr4[(size_t)e * (DDIM / 4) + it * 32 + lane];
      acc[e] += xv.x * wvv.x + xv.y * wvv.y + xv.z * wvv.z + xv.w * wvv.w;
    }
  }
#pragma unroll
  for (int off = 16; off >= 1; off >>= 1)
#pragma unroll
    for (int e = 0; e < NEXP; ++e) acc[e] += __shfl_xor(acc[e], off, 32);

  if (lane == 0) {
    int b0 = 0;
#pragma unroll
    for (int e = 1; e < NEXP; ++e) if (acc[e] > acc[b0]) b0 = e;
    int b1 = (b0 == 0) ? 1 : 0;
#pragma unroll
    for (int e = 0; e < NEXP; ++e) if (e != b0 && acc[e] > acc[b1]) b1 = e;
    float s0 = acc[b0], s1 = acc[b1];
    float p1 = 1.0f / (1.0f + __expf(s0 - s1));   // softmax over {s0,s1}
    float p0 = 1.0f - p1;
    int r0 = atomicAdd(&counts[b0], 1);
    tokl[b0 * CAP + r0] = t; slotl[b0 * CAP + r0] = 0; wtl[b0 * CAP + r0] = p0;
    int r1 = atomicAdd(&counts[b1], 1);
    tokl[b1 * CAP + r1] = t; slotl[b1 * CAP + r1] = 1; wtl[b1 * CAP + r1] = p1;
  }
}

// ---------------------------------------------------------------- K2: gate/up
// Block = 128 threads (4 waves). Tile: 128 gathered tokens x 64 F columns,
// each wave owns 32 rows x 64 cols (2 A frags x 4 B frags -> 16 WMMA/chunk).
__global__ __launch_bounds__(128) void moe_k2_gateup(
    const float* __restrict__ x, const float* __restrict__ wg,
    const float* __restrict__ wu, const int* __restrict__ counts,
    const int* __restrict__ tokl, const float* __restrict__ wtl,
    __bf16* __restrict__ hid) {
  __shared__ __bf16 As[128 * LST];
  __shared__ __bf16 Bg[64 * LST];
  __shared__ __bf16 Bu[64 * LST];
  __shared__ int    rows_s[128];
  __shared__ float  wts_s[128];

  const int e    = blockIdx.z;
  const int cnt  = counts[e];
  const int row0 = blockIdx.y * 128;
  if (row0 >= cnt) return;                 // uniform exit, EXEC stays full below
  const int f0 = blockIdx.x * 64;
  int nvalid = cnt - row0; if (nvalid > 128) nvalid = 128;

  const int tid = threadIdx.x;
  if (tid < nvalid) {
    rows_s[tid] = tokl[e * CAP + row0 + tid];
    wts_s[tid]  = wtl[e * CAP + row0 + tid];
  } else { rows_s[tid] = -1; wts_s[tid] = 0.0f; }
  __syncthreads();

  const int bn = tid >> 1;                 // 0..63: B column
  const int lh = tid & 1;                  // half of the 32-wide K chunk
  const int lane = tid & 31;
  const int wv   = tid >> 5;               // wave id 0..3 -> 32-row sub-tile
  const int m16  = lane & 15;
  const int hi   = lane >> 4;

  v8f accG[2][4], accU[2][4];
#pragma unroll
  for (int g = 0; g < 2; ++g)
#pragma unroll
    for (int s = 0; s < 4; ++s) { accG[g][s] = vzero8(); accU[g][s] = vzero8(); }

  const int trow = rows_s[tid];
  const size_t xbase = (size_t)(trow < 0 ? 0 : trow) * DDIM;
  const size_t wbase = ((size_t)(e * FDIM + f0 + bn)) * DDIM;

  for (int k0 = 0; k0 < DDIM; k0 += 32) {
    // A: gathered x rows (one full 32-wide row per thread), f32 -> bf16
    __bf16* da = &As[tid * LST];
    if (trow >= 0) {
      cvt16_f32_bf16(x + xbase + k0, da);
      cvt16_f32_bf16(x + xbase + k0 + 16, da + 16);
    } else {
#pragma unroll
      for (int j = 0; j < 32; ++j) da[j] = (__bf16)0.0f;
    }
    // B: gate & up weights (row per F col, contiguous along D), f32 -> bf16
    cvt16_f32_bf16(wg + wbase + k0 + lh * 16, &Bg[bn * LST + lh * 16]);
    cvt16_f32_bf16(wu + wbase + k0 + lh * 16, &Bu[bn * LST + lh * 16]);
    if (k0 + 32 < DDIM) {
      __builtin_prefetch(wg + wbase + k0 + 32, 0, 3);   // global_prefetch_b8
      __builtin_prefetch(wu + wbase + k0 + 32, 0, 3);
      if (trow >= 0) __builtin_prefetch(x + xbase + k0 + 32, 0, 3);
    }
    __syncthreads();

    ABf16 a0 = load_afrag(&As[(wv * 32 + m16) * LST], hi);
    ABf16 a1 = load_afrag(&As[(wv * 32 + 16 + m16) * LST], hi);
#pragma unroll
    for (int s = 0; s < 4; ++s) {
      ABf16 bg = load_bfrag(&Bg[(s * 16 + m16) * LST], hi);
      ABf16 bu = load_bfrag(&Bu[(s * 16 + m16) * LST], hi);
      accG[0][s] = wmma_bf16(a0, bg, accG[0][s]);
      accG[1][s] = wmma_bf16(a1, bg, accG[1][s]);
      accU[0][s] = wmma_bf16(a0, bu, accU[0][s]);
      accU[1][s] = wmma_bf16(a1, bu, accU[1][s]);
    }
    __syncthreads();
  }

  // Epilogue: SwiGLU, scale by routing prob, store bf16 hidden (compacted rows)
#pragma unroll
  for (int g = 0; g < 2; ++g)
#pragma unroll
    for (int s = 0; s < 4; ++s)
#pragma unroll
      for (int r = 0; r < 8; ++r) {
        const int tr = wv * 32 + g * 16 + r + hi * 8;  // C/D: lanes 16-31 -> M+8
        if (tr < nvalid) {
          float gg = accG[g][s][r], uu = accU[g][s][r];
          float h = (gg / (1.0f + __expf(-gg))) * uu * wts_s[tr];
          hid[((size_t)e * CAP + row0 + tr) * FDIM + f0 + s * 16 + m16] = (__bf16)h;
        }
      }
}

// ---------------------------------------------------------------- K3: down proj
// Block = 128 threads. Tile: 128 compacted rows x 64 D columns, K = F = 768.
// A tile (bf16 hidden, contiguous rows) is DMA'd by the Tensor Data Mover with
// LDS padding matching the 80B row stride; B tile is converted f32->bf16 by VALU.
__global__ __launch_bounds__(128) void moe_k3_down(
    const __bf16* __restrict__ hid, const float* __restrict__ wd,
    const int* __restrict__ counts, const int* __restrict__ tokl,
    const int* __restrict__ slotl, float* __restrict__ part) {
  __shared__ __bf16 As[128 * LST];
  __shared__ __bf16 Bs[64 * LST];
  __shared__ int rows_s[128];
  __shared__ int slots_s[128];

  const int e    = blockIdx.z;
  const int cnt  = counts[e];
  const int row0 = blockIdx.y * 128;
  if (row0 >= cnt) return;
  const int d0 = blockIdx.x * 64;
  int nvalid = cnt - row0; if (nvalid > 128) nvalid = 128;

  const int tid = threadIdx.x;
  if (tid < nvalid) {
    rows_s[tid]  = tokl[e * CAP + row0 + tid];
    slots_s[tid] = slotl[e * CAP + row0 + tid];
  } else { rows_s[tid] = 0; slots_s[tid] = 0; }

  const int bn = tid >> 1;
  const int lh = tid & 1;
  const int lane = tid & 31;
  const int wv   = tid >> 5;
  const int m16  = lane & 15;
  const int hi   = lane >> 4;

  v8f acc[2][4];
#pragma unroll
  for (int g = 0; g < 2; ++g)
#pragma unroll
    for (int s = 0; s < 4; ++s) acc[g][s] = vzero8();

  // TDM descriptor (D#) constants for the A tile: 2D, 32 x 128 rows of bf16,
  // row stride FDIM elements, LDS pad 4 DWORDs after every 16 DWORDs (-> 80B rows).
  const unsigned int lds_base = (unsigned int)(size_t)&As[0];
  const size_t gbase = (size_t)hid + ((size_t)e * CAP + row0) * FDIM * 2;
  i32x8 g1;
  g1[0] = (1 << 16)          // data_size = 2 bytes
        | (1 << 20)          // pad_enable
        | (3 << 22)          // pad_interval code 3 -> every 16 DWORDs (64B)
        | (3 << 25);         // pad_amount  code 3 -> 4 DWORDs (16B)
  g1[1] = 0;                 // atomic barrier addr, tensor_dim0 low16 (dim0=2^30)
  g1[2] = 0x4000;            // tensor_dim0 hi16 = 0x4000, tensor_dim1 low16 = 0
  g1[3] = 0x4000 | (32 << 16);   // tensor_dim1 hi16, tile_dim0 = 32
  g1[4] = 128;               // tile_dim1 = 128 rows, tile_dim2 = 0
  g1[5] = FDIM;              // tensor_dim0_stride low32
  g1[6] = 0;
  g1[7] = 0;
  i32x4 gz4;
  gz4[0] = 0; gz4[1] = 0; gz4[2] = 0; gz4[3] = 0;
  i32x8 gz8;
#pragma unroll
  for (int i = 0; i < 8; ++i) gz8[i] = 0;

  const size_t wbase = ((size_t)e * DDIM + d0 + bn) * FDIM;   // B row (f32)

  for (int k0 = 0; k0 < FDIM; k0 += 32) {
    // A: Tensor Data Mover DMA into padded LDS tile (one issue per workgroup).
    if (wv == 0) {
      const size_t ga = gbase + (size_t)k0 * 2;
      u32x4 g0;
      g0[0] = 1u;                                  // count = 1 valid descriptor
      g0[1] = lds_base;                            // LDS byte address
      g0[2] = (unsigned int)ga;                    // global_addr[31:0]
      g0[3] = ((unsigned int)(ga >> 32) & 0x1FFFFFFu) | (2u << 30);  // type=2
      __builtin_amdgcn_tensor_load_to_lds(g0, g1, gz4, gz4, gz8, 0);
    }
    // B: down-proj weights f32 -> bf16 (row per D col, contiguous along F)
    cvt16_f32_bf16(wd + wbase + k0 + lh * 16, &Bs[bn * LST + lh * 16]);
    if (k0 + 32 < FDIM) __builtin_prefetch(wd + wbase + k0 + 32, 0, 3);
    if (wv == 0) __builtin_amdgcn_s_wait_tensorcnt(0);
    __syncthreads();

    ABf16 a0 = load_afrag(&As[(wv * 32 + m16) * LST], hi);
    ABf16 a1 = load_afrag(&As[(wv * 32 + 16 + m16) * LST], hi);
#pragma unroll
    for (int s = 0; s < 4; ++s) {
      ABf16 b = load_bfrag(&Bs[(s * 16 + m16) * LST], hi);
      acc[0][s] = wmma_bf16(a0, b, acc[0][s]);
      acc[1][s] = wmma_bf16(a1, b, acc[1][s]);
    }
    __syncthreads();
  }

#pragma unroll
  for (int g = 0; g < 2; ++g)
#pragma unroll
    for (int s = 0; s < 4; ++s)
#pragma unroll
      for (int r = 0; r < 8; ++r) {
        const int tr = wv * 32 + g * 16 + r + hi * 8;
        if (tr < nvalid) {
          const int t  = rows_s[tr];
          const int sl = slots_s[tr];
          part[((size_t)t * 2 + sl) * DDIM + d0 + s * 16 + m16] = acc[g][s][r];
        }
      }
}

// ---------------------------------------------------------------- K4: combine
__global__ __launch_bounds__(256) void moe_k4_combine(
    const float4* __restrict__ p4, float4* __restrict__ out4) {
  const size_t i = (size_t)blockIdx.x * 256 + threadIdx.x;   // over NTOK*DDIM/4
  const size_t t = i / (DDIM / 4);
  const size_t j = i % (DDIM / 4);
  float4 a = p4[(t * 2 + 0) * (DDIM / 4) + j];
  float4 b = p4[(t * 2 + 1) * (DDIM / 4) + j];
  out4[i] = make_float4(a.x + b.x, a.y + b.y, a.z + b.z, a.w + b.w);
}

// ---------------------------------------------------------------- launch
// Workspace layout (bytes, 256-aligned):
static const size_t OFF_COUNTS = 0;
static const size_t OFF_TOK    = 256;
static const size_t OFF_SLOT   = OFF_TOK  + (size_t)NEXP * CAP * 4;      //  64 KB
static const size_t OFF_WT     = OFF_SLOT + (size_t)NEXP * CAP * 4;      //  64 KB
static const size_t OFF_HID    = OFF_WT   + (size_t)NEXP * CAP * 4;      //  64 KB
static const size_t OFF_PART   = OFF_HID  + (size_t)NEXP * CAP * FDIM * 2; // 24 MB
// total = OFF_PART + NTOK*2*DDIM*4  (~56.2 MB)

extern "C" void kernel_launch(void* const* d_in, const int* in_sizes, int n_in,
                              void* d_out, int out_size, void* d_ws, size_t ws_size,
                              hipStream_t stream) {
  (void)in_sizes; (void)n_in; (void)out_size; (void)ws_size;
  const float* x  = (const float*)d_in[0];
  const float* wr = (const float*)d_in[1];
  const float* wg = (const float*)d_in[2];
  const float* wu = (const float*)d_in[3];
  const float* wd = (const float*)d_in[4];
  float* out = (float*)d_out;

  char* ws = (char*)d_ws;
  int*    counts = (int*)   (ws + OFF_COUNTS);
  int*    tokl   = (int*)   (ws + OFF_TOK);
  int*    slotl  = (int*)   (ws + OFF_SLOT);
  float*  wtl    = (float*) (ws + OFF_WT);
  __bf16* hid    = (__bf16*)(ws + OFF_HID);
  float*  part   = (float*) (ws + OFF_PART);

  moe_k0_init<<<1, 32, 0, stream>>>(counts);
  moe_k1_router<<<NTOK / 8, 256, 0, stream>>>(x, wr, counts, tokl, slotl, wtl);
  moe_k2_gateup<<<dim3(FDIM / 64, CAP / 128, NEXP), 128, 0, stream>>>(
      x, wg, wu, counts, tokl, wtl, hid);
  moe_k3_down<<<dim3(DDIM / 64, CAP / 128, NEXP), 128, 0, stream>>>(
      hid, wd, counts, tokl, slotl, part);
  moe_k4_combine<<<(NTOK * DDIM / 4) / 256, 256, 0, stream>>>(
      (const float4*)part, (float4*)out);
}
